// GCNet_64922725646765
// MI455X (gfx1250) — matrix operand order repaired and checked by reading, not compiled, not run on hardware
//
#include <hip/hip_runtime.h>
#include <hip/hip_bf16.h>
#include <math.h>

// ---------------------------------------------------------------------------
// GCN (2-layer, PyG GCNConv semantics) for MI455X / gfx1250, wave32 + WMMA.
//   h  = relu( Agg( (x@W1), norm ) + b1 )        HID = 16
//   o  = log_softmax( Agg( (h@W2), norm ) + b2 ) NC  = 40
// norm(e) = rsqrt(deg[src]) * rsqrt(deg[dst]),  deg includes self-loops.
// Self-loop contribution handled analytically (no extra atomic edges).
// ---------------------------------------------------------------------------

#define K1        1433
#define HID       16
#define NC        40
#define KSTEPS1   45          // ceil(1433/32); last step (t=44) is peeled

typedef __attribute__((ext_vector_type(16))) _Float16 v16h;
typedef __attribute__((ext_vector_type(2)))  __fp16   fp16x2;   // cvt_pkrtz result type
typedef __attribute__((ext_vector_type(8)))  float    v8f;

union Frag {                  // 32 bytes = 8 VGPRs per lane
    v16h          v;
    uint4         q[2];
    unsigned int  u[8];
    _Float16      h[16];
};

// one v_cvt_pk_f16_f32 instead of two cvt + pack
static __device__ inline unsigned int pk2(float a, float b) {
    union { fp16x2 h; unsigned int u; } c;
    c.h = __builtin_amdgcn_cvt_pkrtz(a, b);
    return c.u;
}

// ------------------------------ utility ------------------------------------
__global__ void k_fill0(float* __restrict__ p, long long n) {
    long long i = (long long)blockIdx.x * blockDim.x + threadIdx.x;
    if (i < n) p[i] = 0.0f;
}

__global__ void k_deg(const long long* __restrict__ dst, float* __restrict__ deg, int E) {
    int e = blockIdx.x * blockDim.x + threadIdx.x;
    if (e < E) atomicAdd(&deg[dst[e]], 1.0f);
}

__global__ void k_dinv(float* __restrict__ deg, int n) {
    int i = blockIdx.x * blockDim.x + threadIdx.x;
    if (i < n) deg[i] = rsqrtf(deg[i] + 1.0f);   // +1 = self-loop; always > 0
}

// ------------------------------ GEMM1 (WMMA) -------------------------------
// xw[N,16] = x[N,1433] @ W1[1433,16]; f32 stream -> f16 WMMA, f32 accumulate.
// W1 staged in LDS pre-swizzled into B-fragment layout (ISA 7.12.2):
//   dword [t][l][j] packs K = t*32 + (l<16?0:8) + 2(j&3) + (j<4?0:16)(+1), column n = l&15.
__global__ __launch_bounds__(128) void k_gemm1(const float* __restrict__ x,
                                               const float* __restrict__ W1,
                                               float* __restrict__ xw,
                                               int nstrips) {
    __shared__ unsigned int bfrag[KSTEPS1 * 32 * 8];   // 46080 B
    for (int d = threadIdx.x; d < KSTEPS1 * 32 * 8; d += blockDim.x) {
        int j = d & 7, l = (d >> 3) & 31, t = d >> 8;
        int n  = l & 15;
        int k0 = t * 32 + ((l & 16) ? 8 : 0) + 2 * (j & 3) + ((j & 4) ? 16 : 0);
        float f0 = (k0     < K1) ? W1[k0 * HID + n]       : 0.0f;
        float f1 = (k0 + 1 < K1) ? W1[(k0 + 1) * HID + n] : 0.0f;
        bfrag[d] = pk2(f0, f1);
    }
    __syncthreads();

    const int lane   = threadIdx.x & 31;
    const int wv     = threadIdx.x >> 5;
    const int nwaves = (gridDim.x * blockDim.x) >> 5;
    const int loff   = (lane & 16) ? 8 : 0;            // per-lane K offset (A layout)
    const unsigned int* lbase = &bfrag[lane * 8];      // per-lane B-frag base

    for (int strip = blockIdx.x * (blockDim.x >> 5) + wv; strip < nstrips; strip += nwaves) {
        const int    row = strip * 16 + (lane & 15);
        const float* xr  = x + (size_t)row * K1 + loff;  // lane base; imm offsets below
        v8f acc = {0.f, 0.f, 0.f, 0.f, 0.f, 0.f, 0.f, 0.f};

        // main loop: every index < 1433 for all lanes (max 43*32+8+23 = 1407)
#pragma unroll 4
        for (int t = 0; t < KSTEPS1 - 1; ++t) {
            Frag a, b;
            const int kb = t * 32;                     // + imm element offsets
#pragma unroll
            for (int j = 0; j < 4; ++j) {
                float f0 = __builtin_nontemporal_load(xr + kb + 2 * j);
                float f1 = __builtin_nontemporal_load(xr + kb + 2 * j + 1);
                float f2 = __builtin_nontemporal_load(xr + kb + 16 + 2 * j);
                float f3 = __builtin_nontemporal_load(xr + kb + 16 + 2 * j + 1);
                a.u[j]     = pk2(f0, f1);
                a.u[j + 4] = pk2(f2, f3);
            }
            const uint4* bp = (const uint4*)(lbase + t * 256);
            b.q[0] = bp[0];
            b.q[1] = bp[1];
            acc = __builtin_amdgcn_wmma_f32_16x16x32_f16(false, a.v, false, b.v,
                                                         (short)0, acc, false, false);
        }

        // peeled tail t = 44: K in [1408,1440); only lanes>=16, high chunk,
        // elements 9..15 exceed K1-1=1432 (element 8 there is exactly K=1432).
        {
            const int kb = (KSTEPS1 - 1) * 32;         // 1408
            Frag a, b;
#pragma unroll
            for (int j = 0; j < 4; ++j)
                a.u[j] = pk2(xr[kb + 2 * j], xr[kb + 2 * j + 1]);
            if (loff == 0) {                            // lanes 0-15: K 1424..1431 valid
#pragma unroll
                for (int j = 0; j < 4; ++j)
                    a.u[j + 4] = pk2(xr[kb + 16 + 2 * j], xr[kb + 16 + 2 * j + 1]);
            } else {                                    // lanes 16-31: only K=1432 valid
                a.u[4] = pk2(xr[kb + 16], 0.0f);
                a.u[5] = 0u; a.u[6] = 0u; a.u[7] = 0u;
            }
            const uint4* bp = (const uint4*)(lbase + (KSTEPS1 - 1) * 256);
            b.q[0] = bp[0];
            b.q[1] = bp[1];
            acc = __builtin_amdgcn_wmma_f32_16x16x32_f16(false, a.v, false, b.v,
                                                         (short)0, acc, false, false);
        }

        // C/D layout: VGPR j -> M = j (+8 for lanes 16-31), N = lane&15.
        // Single base address; j-stride is HID floats = 64 B -> imm offsets.
        {
            const int m0  = (lane & 16) ? 8 : 0;
            float* op = xw + (size_t)(strip * 16 + m0) * HID + (lane & 15);
#pragma unroll
            for (int j = 0; j < 8; ++j)
                op[j * HID] = acc[j];
        }
    }
}

// ------------------------------ GEMM2 (WMMA) -------------------------------
// hw2[N,40] = h[N,16] @ W2[16,40]; K padded to 32 (zeros), N padded to 48.
__global__ __launch_bounds__(128) void k_gemm2(const float* __restrict__ h,
                                               const float* __restrict__ W2,
                                               float* __restrict__ hw2,
                                               int nstrips) {
    __shared__ unsigned int bfrag[3 * 32 * 8];
    for (int d = threadIdx.x; d < 3 * 32 * 8; d += blockDim.x) {
        int j = d & 7, l = (d >> 3) & 31, tile = d >> 8;
        int n  = tile * 16 + (l & 15);
        int k0 = ((l & 16) ? 8 : 0) + 2 * (j & 3) + ((j & 4) ? 16 : 0);
        float f0 = (k0     < HID && n < NC) ? W2[k0 * NC + n]       : 0.0f;
        float f1 = (k0 + 1 < HID && n < NC) ? W2[(k0 + 1) * NC + n] : 0.0f;
        bfrag[d] = pk2(f0, f1);
    }
    __syncthreads();

    const int lane   = threadIdx.x & 31;
    const int wv     = threadIdx.x >> 5;
    const int nwaves = (gridDim.x * blockDim.x) >> 5;
    const unsigned int* lbase = &bfrag[lane * 8];

    for (int strip = blockIdx.x * (blockDim.x >> 5) + wv; strip < nstrips; strip += nwaves) {
        const int     row = strip * 16 + (lane & 15);
        const float4* hr  = (const float4*)(h + (size_t)row * HID + ((lane & 16) ? 8 : 0));
        const float4  p0 = hr[0], p1 = hr[1];
        Frag a;
        a.u[0] = pk2(p0.x, p0.y);
        a.u[1] = pk2(p0.z, p0.w);
        a.u[2] = pk2(p1.x, p1.y);
        a.u[3] = pk2(p1.z, p1.w);
        a.u[4] = 0u; a.u[5] = 0u; a.u[6] = 0u; a.u[7] = 0u;   // K 16..31 zero-pad

#pragma unroll
        for (int tile = 0; tile < 3; ++tile) {
            Frag b;
            const uint4* bp = (const uint4*)(lbase + tile * 256);
            b.q[0] = bp[0];
            b.q[1] = bp[1];
            v8f acc = {0.f, 0.f, 0.f, 0.f, 0.f, 0.f, 0.f, 0.f};
            acc = __builtin_amdgcn_wmma_f32_16x16x32_f16(false, a.v, false, b.v,
                                                         (short)0, acc, false, false);
            const int col = tile * 16 + (lane & 15);
            if (col < NC) {
                const int m0 = (lane & 16) ? 8 : 0;
                float* op = hw2 + (size_t)(strip * 16 + m0) * NC + col;   // j-stride 160 B
#pragma unroll
                for (int j = 0; j < 8; ++j)
                    op[j * NC] = acc[j];
            }
        }
    }
}

// ------------------------------ edge scatter -------------------------------
// out[dst] += feat[src] * dinv[src]*dinv[dst]; NCOMP4 float4-groups per edge.
// NCOMP4 compile-time so tid/NCOMP4 is a mul-shift, not an integer divide.
template <int NCOMP4, int ROWLEN>
__global__ void k_scatter(const long long* __restrict__ src,
                          const long long* __restrict__ dst,
                          const float* __restrict__ dinv,
                          const float* __restrict__ feat,
                          float* __restrict__ out,
                          int E) {
    long long tid   = (long long)blockIdx.x * blockDim.x + threadIdx.x;
    long long total = (long long)E * NCOMP4;
    if (tid >= total) return;
    int e = (int)(tid / NCOMP4);
    int g = (int)(tid % NCOMP4);
    long long s = src[e], d = dst[e];
    float w = dinv[s] * dinv[d];
    const float4 v = *(const float4*)(feat + (size_t)s * ROWLEN + 4 * g);
    float* o = out + (size_t)d * ROWLEN + 4 * g;
    atomicAdd(o + 0, v.x * w);
    atomicAdd(o + 1, v.y * w);
    atomicAdd(o + 2, v.z * w);
    atomicAdd(o + 3, v.w * w);
}

// ------------------------------ epilogues ----------------------------------
// layer1: h = relu(agg + xw*dinv^2 (self-loop) + b1), in place over agg.
__global__ void k_post1(float* __restrict__ hagg, const float* __restrict__ xw1,
                        const float* __restrict__ dinv, const float* __restrict__ b1,
                        int n) {
    long long i = (long long)blockIdx.x * blockDim.x + threadIdx.x;
    if (i >= (long long)n * HID) return;
    int node = (int)(i >> 4), j = (int)(i & 15);
    float ds = dinv[node];
    float v  = hagg[i] + xw1[i] * ds * ds + b1[j];
    hagg[i]  = v > 0.0f ? v : 0.0f;
}

// layer2: out = log_softmax(agg + hw2*dinv^2 + b2), in place over out.
__global__ void k_post2(float* __restrict__ out, const float* __restrict__ hw2,
                        const float* __restrict__ dinv, const float* __restrict__ b2,
                        int n) {
    int i = blockIdx.x * blockDim.x + threadIdx.x;
    if (i >= n) return;
    float ds = dinv[i]; ds *= ds;
    float*       o  = out + (size_t)i * NC;
    const float* hh = hw2 + (size_t)i * NC;
    float v[NC];
    float m = -INFINITY;
#pragma unroll
    for (int j = 0; j < NC; ++j) { v[j] = o[j] + hh[j] * ds + b2[j]; m = fmaxf(m, v[j]); }
    float ssum = 0.0f;
#pragma unroll
    for (int j = 0; j < NC; ++j) ssum += expf(v[j] - m);
    float lse = m + logf(ssum);
#pragma unroll
    for (int j = 0; j < NC; ++j) o[j] = v[j] - lse;
}

// ------------------------------ launcher -----------------------------------
extern "C" void kernel_launch(void* const* d_in, const int* in_sizes, int n_in,
                              void* d_out, int out_size, void* d_ws, size_t ws_size,
                              hipStream_t stream) {
    const float*     x   = (const float*)d_in[0];
    const long long* ei  = (const long long*)d_in[1];   // [2,E] int64
    const float*     W1  = (const float*)d_in[2];
    const float*     b1  = (const float*)d_in[3];
    const float*     W2  = (const float*)d_in[4];
    const float*     b2  = (const float*)d_in[5];
    float*           out = (float*)d_out;

    const int n = in_sizes[0] / K1;      // 100000
    const int E = in_sizes[1] / 2;       // 3200000
    const long long* src = ei;
    const long long* dst = ei + E;

    // workspace: deg/dinv [n] | xw1 [n*16] | agg1/h [n*16] | hw2 [n*40]
    float* deg  = (float*)d_ws;
    float* xw1  = deg + n;
    float* hbuf = xw1 + (size_t)n * HID;
    float* hw2  = hbuf + (size_t)n * HID;

    const int nstrips = n / 16;          // 6250

    // zero aggregation targets (harness poisons once; we re-zero every call)
    k_fill0<<<(n + 255) / 256, 256, 0, stream>>>(deg, n);
    k_fill0<<<((long long)n * HID + 255) / 256, 256, 0, stream>>>(hbuf, (long long)n * HID);
    k_fill0<<<((long long)n * NC + 255) / 256, 256, 0, stream>>>(out, (long long)n * NC);

    // degrees -> dinv (in place)
    k_deg<<<(E + 255) / 256, 256, 0, stream>>>(dst, deg, E);
    k_dinv<<<(n + 255) / 256, 256, 0, stream>>>(deg, n);

    // layer 1
    k_gemm1<<<(nstrips + 3) / 4, 128, 0, stream>>>(x, W1, xw1, nstrips);
    k_scatter<HID / 4, HID><<<(unsigned)(((long long)E * (HID / 4) + 255) / 256), 256, 0, stream>>>(
        src, dst, deg, xw1, hbuf, E);
    k_post1<<<(unsigned)(((long long)n * HID + 255) / 256), 256, 0, stream>>>(
        hbuf, xw1, deg, b1, n);

    // layer 2
    k_gemm2<<<(nstrips + 3) / 4, 128, 0, stream>>>(hbuf, W2, hw2, nstrips);
    k_scatter<NC / 4, NC><<<(unsigned)(((long long)E * (NC / 4) + 255) / 256), 256, 0, stream>>>(
        src, dst, deg, hw2, out, E);
    k_post2<<<(n + 255) / 256, 256, 0, stream>>>(out, hw2, deg, b2, n);
}
